// GraphBackbone_32401233281333
// MI455X (gfx1250) — matrix operand order repaired and checked by reading, not compiled, not run on hardware
//
#include <hip/hip_runtime.h>

#define HID    64
#define IN_DIM 37
#define BN_EPS 1e-5f

typedef __attribute__((ext_vector_type(16))) _Float16 v16h;
typedef __attribute__((ext_vector_type(8)))  float    v8f;
typedef __attribute__((ext_vector_type(4)))  float    v4f;

// ---------------- padding pre-passes ----------------

// xpad[n][k] = (k < 37) ? x[n*37+k] : 0     (N x 64, 16B-aligned rows)
__global__ void xpad_kernel(const float* __restrict__ x, float* __restrict__ xpad, int N) {
    int t = blockIdx.x * blockDim.x + threadIdx.x;
    int n = t >> 6, k = t & 63;
    if (n < N) xpad[t] = (k < IN_DIM) ? x[(size_t)n * IN_DIM + k] : 0.0f;
}

// wt[l][n][k] = (k < K_l) ? W_l[k*64+n] : 0  (transposed + zero-padded weights)
__global__ void wt_kernel(const float* __restrict__ W1, const float* __restrict__ W2,
                          const float* __restrict__ W3, float* __restrict__ wt) {
    int t = blockIdx.x * blockDim.x + threadIdx.x;   // 3*64*64 threads
    if (t >= 3 * HID * HID) return;
    int l = t >> 12;
    int n = (t >> 6) & 63;
    int k = t & 63;
    const float* W = (l == 0) ? W1 : (l == 1) ? W2 : W3;
    int K = (l == 0) ? IN_DIM : HID;
    wt[t] = (k < K) ? W[(size_t)k * HID + n] : 0.0f;
}

// ---------------- degree / normalization ----------------

__global__ void deg_init_kernel(float* deg, int N) {
    int i = blockIdx.x * blockDim.x + threadIdx.x;
    if (i < N) deg[i] = 1.0f;                     // self-loop contributes 1
}

__global__ void deg_count_kernel(const int* __restrict__ col, float* deg, int E) {
    int e = blockIdx.x * blockDim.x + threadIdx.x;
    if (e < E) atomicAdd(&deg[col[e]], 1.0f);
}

__global__ void dinv_kernel(float* deg, int N) {
    int i = blockIdx.x * blockDim.x + threadIdx.x;
    if (i < N) deg[i] = rsqrtf(deg[i]);           // deg >= 1 always
}

// ---------------- WMMA GEMM: H[N,64] = X[N,64] * WT^T ----------------
// X is N x 64 (padded), WT is 64(col) x 64(k) transposed weights.
// Split-fp16: v = hi + lo -> acc += hi*hi + hi*lo + lo*hi (~22-bit mantissa).
// One wave = one 16x16 tile; 8 waves/block = 32 rows x 64 cols.

__global__ __launch_bounds__(256) void gemm_wmma_kernel(
    const float* __restrict__ X, const float* __restrict__ WT,
    float* __restrict__ H, int N)
{
    const int lane = threadIdx.x & 31;
    const int wave = threadIdx.x >> 5;
    const int half = lane >> 4;       // 0: lanes 0-15, 1: lanes 16-31
    const int nn   = lane & 15;
    const int m0   = blockIdx.x * 32 + (wave >> 2) * 16;
    const int col0 = (wave & 3) * 16;

    int arow = m0 + nn;
    if (arow >= N) arow = N - 1;      // clamp loads; stores are guarded

    const float* pA = X  + (size_t)arow * HID;
    const float* pB = WT + (size_t)(col0 + nn) * HID;

    v8f acc = {};

    #pragma unroll
    for (int chunk = 0; chunk < 2; ++chunk) {
        const int ka0 = chunk * 32 + half * 8;    // A: two 8-float runs
        const int kb0 = chunk * 32 + half * 8;    // B: two 8-float runs

        v4f a0 = *(const v4f*)(pA + ka0);
        v4f a1 = *(const v4f*)(pA + ka0 + 4);
        v4f a2 = *(const v4f*)(pA + ka0 + 16);
        v4f a3 = *(const v4f*)(pA + ka0 + 20);
        v4f b0 = *(const v4f*)(pB + kb0);
        v4f b1 = *(const v4f*)(pB + kb0 + 4);
        v4f b2 = *(const v4f*)(pB + kb0 + 16);
        v4f b3 = *(const v4f*)(pB + kb0 + 20);

        float af[16], bf[16];
        #pragma unroll
        for (int i = 0; i < 4; ++i) {
            af[i]      = a0[i]; af[4 + i]  = a1[i];
            af[8 + i]  = a2[i]; af[12 + i] = a3[i];
            bf[i]      = b0[i]; bf[4 + i]  = b1[i];
            bf[8 + i]  = b2[i]; bf[12 + i] = b3[i];
        }

        v16h a_hi, a_lo, b_hi, b_lo;
        #pragma unroll
        for (int e = 0; e < 16; ++e) {
            _Float16 ah = (_Float16)af[e];
            a_hi[e] = ah;
            a_lo[e] = (_Float16)(af[e] - (float)ah);
            _Float16 bh = (_Float16)bf[e];
            b_hi[e] = bh;
            b_lo[e] = (_Float16)(bf[e] - (float)bh);
        }

        acc = __builtin_amdgcn_wmma_f32_16x16x32_f16(false, a_hi, false, b_lo,
                                                     (short)0, acc, false, false);
        acc = __builtin_amdgcn_wmma_f32_16x16x32_f16(false, a_lo, false, b_hi,
                                                     (short)0, acc, false, false);
        acc = __builtin_amdgcn_wmma_f32_16x16x32_f16(false, a_hi, false, b_hi,
                                                     (short)0, acc, false, false);
    }

    // C/D: VGPR r -> row (half*8 + r), column col0+nn
    float* pH = H + (size_t)(m0 + half * 8) * HID + col0 + nn;
    if (m0 + 16 <= N) {               // fast path: always taken for N%32==0
        #pragma unroll
        for (int r = 0; r < 8; ++r) pH[(size_t)r * HID] = acc[r];
    } else {
        #pragma unroll
        for (int r = 0; r < 8; ++r)
            if (m0 + half * 8 + r < N) pH[(size_t)r * HID] = acc[r];
    }
}

// ---------------- message passing ----------------

// agg[n] = dinv[n]^2 * h[n]  (self-loop term; deterministically overwrites poison)
__global__ void agg_init_kernel(const float* __restrict__ dinv,
                                const float* __restrict__ h,
                                float* __restrict__ agg, int N)
{
    int t = blockIdx.x * blockDim.x + threadIdx.x;
    int node = t >> 6;
    if (node < N) {
        float d = dinv[node];
        agg[t] = d * d * h[t];
    }
}

// 64 threads per edge: coalesced 256B row gather, fp32 atomic scatter (L2-resident)
__global__ void edge_scatter_kernel(const int* __restrict__ row,
                                    const int* __restrict__ col,
                                    const float* __restrict__ dinv,
                                    const float* __restrict__ h,
                                    float* __restrict__ agg, int E)
{
    long long t = (long long)blockIdx.x * blockDim.x + threadIdx.x;
    int e = (int)(t >> 6);
    int c = (int)(t & 63);
    if (e >= E) return;
    int r  = row[e];
    int cl = col[e];
    float w = dinv[r] * dinv[cl];
    atomicAdd(&agg[(size_t)cl * HID + c], w * h[(size_t)r * HID + c]);
}

// bias + BatchNorm(eval) + ReLU + optional residual (in place on xout)
__global__ void post_kernel(const float* __restrict__ agg,
                            const float* __restrict__ bvec,
                            const float* __restrict__ gamma,
                            const float* __restrict__ beta,
                            const float* __restrict__ mean,
                            const float* __restrict__ var,
                            float* __restrict__ xout, int N, int has_res)
{
    int t = blockIdx.x * blockDim.x + threadIdx.x;
    if (t >= N * HID) return;
    int c = t & 63;
    float v = agg[t] + bvec[c];
    v = (v - mean[c]) * rsqrtf(var[c] + BN_EPS) * gamma[c] + beta[c];
    v = fmaxf(v, 0.0f);
    xout[t] = has_res ? (xout[t] + v) : v;
}

// ---------------- driver ----------------

extern "C" void kernel_launch(void* const* d_in, const int* in_sizes, int n_in,
                              void* d_out, int out_size, void* d_ws, size_t ws_size,
                              hipStream_t stream)
{
    const float* x0    = (const float*)d_in[0];   // [N, 37]
    const int*   eidx  = (const int*)  d_in[1];   // [2, E]
    const float* W1    = (const float*)d_in[2];   // [37, 64]
    const float* W2    = (const float*)d_in[3];   // [64, 64]
    const float* W3    = (const float*)d_in[4];   // [64, 64]
    const float* bv    = (const float*)d_in[5];   // [3, 64]
    const float* gam   = (const float*)d_in[6];
    const float* bet   = (const float*)d_in[7];
    const float* rmean = (const float*)d_in[8];
    const float* rvar  = (const float*)d_in[9];

    const int N = in_sizes[0] / IN_DIM;
    const int E = in_sizes[1] / 2;
    const int* row = eidx;
    const int* col = eidx + E;

    float* xcur = (float*)d_out;                       // [N,64] running features
    float* ws   = (float*)d_ws;
    float* h1   = ws;                                  // [N*64]
    float* agg  = ws + (size_t)N * HID;                // [N*64]
    float* xpad = ws + (size_t)2 * N * HID;            // [N*64]
    float* dinv = ws + (size_t)3 * N * HID;            // [N]
    float* wt   = dinv + N;                            // [3*64*64]

    dim3 blk(256);
    int gN  = (N + 255) / 256;
    int gE  = (E + 255) / 256;
    int gNC = (int)(((long long)N * HID + 255) / 256);
    int gEC = (int)(((long long)E * HID + 255) / 256);
    int gMM = (N + 31) / 32;
    int gWT = (3 * HID * HID + 255) / 256;

    xpad_kernel     <<<gNC, blk, 0, stream>>>(x0, xpad, N);
    wt_kernel       <<<gWT, blk, 0, stream>>>(W1, W2, W3, wt);
    deg_init_kernel <<<gN,  blk, 0, stream>>>(dinv, N);
    deg_count_kernel<<<gE,  blk, 0, stream>>>(col, dinv, E);
    dinv_kernel     <<<gN,  blk, 0, stream>>>(dinv, N);

    for (int layer = 0; layer < 3; ++layer) {
        const float* X = (layer == 0) ? xpad : xcur;

        gemm_wmma_kernel   <<<gMM, blk, 0, stream>>>(X, wt + (size_t)layer * HID * HID, h1, N);
        agg_init_kernel    <<<gNC, blk, 0, stream>>>(dinv, h1, agg, N);
        edge_scatter_kernel<<<gEC, blk, 0, stream>>>(row, col, dinv, h1, agg, E);
        post_kernel        <<<gNC, blk, 0, stream>>>(agg,
                                                     bv    + layer * HID,
                                                     gam   + layer * HID,
                                                     bet   + layer * HID,
                                                     rmean + layer * HID,
                                                     rvar  + layer * HID,
                                                     xcur, N, layer > 0 ? 1 : 0);
    }
}
